// Decoder_63720134804045
// MI455X (gfx1250) — matrix operand order repaired and checked by reading, not compiled, not run on hardware
//
#include <hip/hip_runtime.h>
#include <hip/hip_bf16.h>
#include <math.h>

// ---------------------------------------------------------------------------
// Problem constants (from reference): B=32, S=64, D=768, V=50, MAXD=5
// N_NODES=31, N_SLOTS=63, LEAF_DIM=32, LEAF_SLOTS=15, NEIGH=6, PAD_LEN=2880
// ---------------------------------------------------------------------------
#define Bq      32
#define Sq      64
#define DD      768
#define VV      50
#define VPAD    64
#define NNODES  31
#define NSLOTS  63
#define BS      (Bq * Sq)      // 2048
#define PADK    776            // 768 + 8 bf16 pad (16B) per LDS row
#define MT      64             // M tile per workgroup
#define THREADS 256

typedef __attribute__((ext_vector_type(16))) __bf16 v16bf;
typedef __attribute__((ext_vector_type(8)))  __bf16 v8bf;
typedef __attribute__((ext_vector_type(8)))  float  v8f;

__device__ __forceinline__ float gelu_exact(float x) {
    return 0.5f * x * (1.0f + erff(x * 0.70710678118654752f));
}

// A fragment loader (16-bit A 16x32), contiguous K order matching B:
//   lane<16 : M=lane,    K = kb+0..15
//   lane>=16: M=lane-16, K = kb+0..15  (kb pre-offset by caller, +16)
__device__ __forceinline__ v16bf load_a_frag(const __bf16 (*Xs)[PADK],
                                             int row, int kb) {
    v8bf lo = *(const v8bf*)&Xs[row][kb];
    v8bf hi = *(const v8bf*)&Xs[row][kb + 8];
    return __builtin_shufflevector(lo, hi,
            0,1,2,3,4,5,6,7,8,9,10,11,12,13,14,15);
}

// ---------------------------------------------------------------------------
// GEMM helper: X (LDS bf16, 64xD) @ Wt (global bf16, N-major NxD) -> epilogue
// Wave layout: 8 waves = 2 (M groups of 32 rows) x 4 (N groups of NF*16 cols).
// Each wave owns a 32 x (NF*16) tile: 2 A frags, NF B frags, 2*NF WMMAs per
// k-step (4 independent chains for NF=2) -> good ILP, 2x B reuse.
// B loads are software-pipelined 2 k-steps deep so each global B fragment has
// ~8 WMMAs of issue distance to hide L2 latency (only 2 waves/SIMD resident).
//   bias != nullptr : out = gelu(acc + bias[col])
//   mem  != nullptr : out += memory[bs][col] (+ other[bs][col] if oth)
//   Pd   != nullptr : store f32 into Pd (V-logit GEMM), else bf16 -> Hd
// B fragment (32x16 bf16): lane<16 -> N=lane, K=k0..15; lane>=16 -> K=k0+16..31
// N-major weights make each lane's B fragment one contiguous 32B read.
// ---------------------------------------------------------------------------
template<int NF>
__device__ __forceinline__ void gemm_bf16(
    const __bf16 (*Xs)[PADK], const __bf16* __restrict__ Wt,
    __bf16 (*Hd)[PADK], float (*Pd)[VPAD],
    const float* __restrict__ bias,
    const float* __restrict__ mem, const float* __restrict__ oth,
    int nChunks, int m0, int cshift, int tid)
{
    const int lane = tid & 31;
    const int wid  = tid >> 5;
    const int wm   = wid & 1;            // M group (32 rows)
    const int wn   = wid >> 1;           // N group (NF*16 cols)
    const int mr0  = wm * 32 + (lane & 15);
    const int mr1  = mr0 + 16;
    const int kA   = (lane >> 4) << 4;   // +0 or +16
    const int nl   = lane & 15;
    const int kB   = (lane >> 4) << 4;   // +0 or +16
    const int CW   = 4 * NF * 16;        // chunk width (128 or 64)

    for (int nc = 0; nc < nChunks; ++nc) {
        const int nBase = nc * CW;
        v8f acc[2][NF];
#pragma unroll
        for (int mi = 0; mi < 2; ++mi)
#pragma unroll
            for (int ni = 0; ni < NF; ++ni)
                acc[mi][ni] = (v8f){};

        const int n0 = nBase + wn * (NF * 16) + nl;
        const __bf16* wp0 = Wt + (size_t)n0 * DD + kB;

        // ---- depth-2 software pipeline on B fragments ----
        v16bf bq0[NF], bq1[NF];
#pragma unroll
        for (int ni = 0; ni < NF; ++ni)
            bq0[ni] = *(const v16bf*)(wp0 + (size_t)(ni * 16) * DD + 0);
#pragma unroll
        for (int ni = 0; ni < NF; ++ni)
            bq1[ni] = *(const v16bf*)(wp0 + (size_t)(ni * 16) * DD + 32);

#pragma unroll
        for (int k0 = 0; k0 < DD; k0 += 32) {
            // issue k0+64 loads before consuming k0's fragments
            v16bf bn[NF];
            if (k0 + 64 < DD) {
#pragma unroll
                for (int ni = 0; ni < NF; ++ni)
                    bn[ni] = *(const v16bf*)(wp0 + (size_t)(ni * 16) * DD
                                                 + (k0 + 64));
            }
            v16bf a0 = load_a_frag(Xs, mr0, k0 + kA);
            v16bf a1 = load_a_frag(Xs, mr1, k0 + kA);
#pragma unroll
            for (int ni = 0; ni < NF; ++ni) {
                acc[0][ni] = __builtin_amdgcn_wmma_f32_16x16x32_bf16(
                                 false, a0, false, bq0[ni], (short)0,
                                 acc[0][ni], false, false);
                acc[1][ni] = __builtin_amdgcn_wmma_f32_16x16x32_bf16(
                                 false, a1, false, bq0[ni], (short)0,
                                 acc[1][ni], false, false);
            }
            // rotate queue (free after full unroll / register renaming)
#pragma unroll
            for (int ni = 0; ni < NF; ++ni) {
                bq0[ni] = bq1[ni];
                bq1[ni] = bn[ni];
            }
        }
        // epilogue: C layout — lane l, vgpr j -> row j+(l>=16?8:0), col l&15
        const int rb = (lane >> 4) << 3;
#pragma unroll
        for (int mi = 0; mi < 2; ++mi) {
#pragma unroll
            for (int ni = 0; ni < NF; ++ni) {
                const int col = nBase + wn * (NF * 16) + ni * 16 + nl;
#pragma unroll
                for (int j = 0; j < 8; ++j) {
                    const int lm = wm * 32 + mi * 16 + rb + j;
                    float v = acc[mi][ni][j];
                    if (bias) v = gelu_exact(v + bias[col]);
                    if (mem) {
                        const int bs = (m0 + lm) >> cshift;
                        v += mem[(size_t)bs * DD + col];
                        if (oth) v += oth[(size_t)bs * DD + col];
                    }
                    if (Pd) Pd[lm][col] = v;
                    else    Hd[lm][col] = (__bf16)v;
                }
            }
        }
    }
}

// LayerNorm of 64 rows (bf16 in Hs) -> ln(g,b) -> bf16 in Xd.
// Wave w handles rows 8w..8w+7; wave32 shuffle reduction.
__device__ __forceinline__ void layernorm_rows(
    const __bf16 (*Hs)[PADK], __bf16 (*Xd)[PADK],
    const float* __restrict__ g, const float* __restrict__ bta, int tid)
{
    const int lane = tid & 31;
    const int wid  = tid >> 5;
#pragma unroll
    for (int rr = 0; rr < 8; ++rr) {
        const int row = wid * 8 + rr;
        float s = 0.f, sq = 0.f;
        for (int c = lane; c < DD; c += 32) {
            float v = (float)Hs[row][c];
            s += v; sq += v * v;
        }
#pragma unroll
        for (int off = 16; off > 0; off >>= 1) {
            s  += __shfl_xor(s,  off, 32);
            sq += __shfl_xor(sq, off, 32);
        }
        const float mean = s * (1.0f / DD);
        const float var  = sq * (1.0f / DD) - mean * mean;
        const float rstd = rsqrtf(var + 1e-5f);
        for (int c = lane; c < DD; c += 32) {
            float v = (float)Hs[row][c];
            Xd[row][c] = (__bf16)(((v - mean) * rstd) * g[c] + bta[c]);
        }
    }
}

// ---------------------------------------------------------------------------
// Fused per-depth kernel: 64 token-node rows per workgroup, full MLP chain.
// ---------------------------------------------------------------------------
__global__ __launch_bounds__(THREADS)
void fused_depth(const float* __restrict__ feat_embs,
                 const int*   __restrict__ feat_idx,
                 const float* __restrict__ memory,
                 const float* __restrict__ other,
                 const float* __restrict__ exm,
                 const __bf16* __restrict__ W1t, const float* __restrict__ b1,
                 const __bf16* __restrict__ W2t, const float* __restrict__ b2,
                 const __bf16* __restrict__ W3t, const float* __restrict__ b3,
                 const __bf16* __restrict__ Woutb,
                 const float* __restrict__ ln_g, const float* __restrict__ ln_b,
                 float* __restrict__ out,
                 int lo, int cshift, int useOther)
{
    __shared__ __bf16 Xb[MT][PADK];     // 99,328 B
    __shared__ __bf16 Hb[MT][PADK];     // 99,328 B
    __shared__ float  P [MT][VPAD];     // 16,384 B   (total ~215 KB of 320 KB)

    const int tid   = threadIdx.x;
    const int m0    = blockIdx.x * MT;
    const int cmask = (1 << cshift) - 1;

    // Gather embedding rows (f32 -> bf16) into LDS
    for (int idx = tid; idx < MT * DD; idx += THREADS) {
        const int lr = idx / DD;
        const int c  = idx - lr * DD;
        const int r  = m0 + lr;
        const int bs = r >> cshift;
        const int node = lo + (r & cmask);
        const int fi = feat_idx[bs * NNODES + node];
        Xb[lr][c] = (__bf16)feat_embs[(size_t)fi * DD + c];
    }
    __syncthreads();

    // h = gelu(emb@W1 + b1) + memory (+ other)
    gemm_bf16<2>(Xb, W1t, Hb, nullptr, b1, memory,
                 useOther ? other : nullptr, DD / 128, m0, cshift, tid);
    __syncthreads();
    layernorm_rows(Hb, Xb, ln_g, ln_b, tid);
    __syncthreads();
    // x = gelu(x@W2 + b2)
    gemm_bf16<2>(Xb, W2t, Hb, nullptr, b2, nullptr, nullptr,
                 DD / 128, m0, cshift, tid);
    __syncthreads();
    layernorm_rows(Hb, Xb, ln_g, ln_b, tid);
    __syncthreads();
    // x = gelu(x@W3 + b3)
    gemm_bf16<2>(Xb, W3t, Hb, nullptr, b3, nullptr, nullptr,
                 DD / 128, m0, cshift, tid);
    __syncthreads();
    // logits = x @ Wout  (V padded to 64, pad cols produce zeros -> ignored)
    gemm_bf16<1>(Hb, Woutb, nullptr, P, nullptr, nullptr, nullptr,
                 1, m0, cshift, tid);
    __syncthreads();

    // softmax over 50 valid columns, scale by ex mask, store
    if (tid < MT) {
        const int lm   = tid;
        const int r    = m0 + lm;
        const int bs   = r >> cshift;
        const int node = lo + (r & cmask);
        const float mask = exm[bs * NNODES + node];
        float mx = -3.4e38f;
        for (int c = 0; c < VV; ++c) mx = fmaxf(mx, P[lm][c]);
        float ssum = 0.f;
        for (int c = 0; c < VV; ++c) {
            float e = expf(P[lm][c] - mx);
            P[lm][c] = e;
            ssum += e;
        }
        const float scale = mask / ssum;
        float* op = out + ((size_t)bs * NSLOTS + node) * VV;
        for (int c = 0; c < VV; ++c) op[c] = P[lm][c] * scale;
    }
}

// ---------------------------------------------------------------------------
// Prep kernels
// ---------------------------------------------------------------------------
// Wt[n*D + k] = (n < N) ? W[k*N + n] : 0   (f32 -> bf16, transpose to N-major)
__global__ void transpose_to_bf16(const float* __restrict__ W,
                                  __bf16* __restrict__ Wt, int N, int Npad)
{
    const int idx = blockIdx.x * blockDim.x + threadIdx.x;
    if (idx >= Npad * DD) return;
    const int n = idx / DD;
    const int k = idx - n * DD;
    const float v = (n < N) ? W[(size_t)k * N + n] : 0.f;
    Wt[idx] = (__bf16)v;
}

// ex-mask tree: ex[0]=tok_valid; ex[i]=ex[p] & (tgt[p] in {0,1}), p=(i-1)/2
__global__ void build_ex(const int* __restrict__ tgt,
                         const int* __restrict__ seq_length,
                         float* __restrict__ exm)
{
    const int t = blockIdx.x * blockDim.x + threadIdx.x;
    if (t >= BS) return;
    const int b = t >> 6;
    const int s = t & 63;
    bool e[NNODES];
    e[0] = s < seq_length[b];
    const int* tg = tgt + (size_t)t * NSLOTS;
    for (int i = 1; i < NNODES; ++i) {
        const int p  = (i - 1) >> 1;
        const int tp = tg[p];
        e[i] = e[p] && (tp == 0 || tp == 1);
    }
    for (int i = 0; i < NNODES; ++i) exm[t * NNODES + i] = e[i] ? 1.f : 0.f;
}

// other_leaves for depth d: sparse (<=5*cnt active 32-row blocks of leaf_W).
// One workgroup per (b,s); thread t owns columns t, t+256, t+512.
// Offsets replicate reference quirk: [-3,-2,-1,1,2] (neighbor slot 5 unused).
__global__ __launch_bounds__(THREADS)
void build_other(const int* __restrict__ tgt, const float* __restrict__ exm,
                 const float* __restrict__ leaf_emb,
                 const float* __restrict__ leaf_W,
                 const float* __restrict__ leaf_b,
                 float* __restrict__ other, int a, int cntPrev)
{
    const int bs  = blockIdx.x;
    const int b   = bs >> 6;
    const int s   = bs & 63;
    const int tid = threadIdx.x;
    float acc0 = leaf_b[tid];
    float acc1 = leaf_b[tid + 256];
    float acc2 = leaf_b[tid + 512];
    const int offs[5] = {-3, -2, -1, 1, 2};
    for (int n = 0; n < 5; ++n) {
        const int s2 = s + offs[n];
        if (s2 < 0 || s2 >= Sq) continue;
        const int base = (b << 6) + s2;
        for (int slot = 0; slot < cntPrev; ++slot) {
            if (exm[base * NNODES + a + slot] == 0.f) continue;
            const int label = tgt[(size_t)base * NSLOTS + a + slot];
            const float* ep = leaf_emb + label * 32;
            const float* wr = leaf_W + (size_t)((n * 15 + slot) * 32) * DD;
            for (int k = 0; k < 32; ++k) {
                const float ev = ep[k];
                const float* w = wr + (size_t)k * DD;
                acc0 += ev * w[tid];
                acc1 += ev * w[tid + 256];
                acc2 += ev * w[tid + 512];
            }
        }
    }
    float* op = other + (size_t)bs * DD;
    op[tid]       = acc0;
    op[tid + 256] = acc1;
    op[tid + 512] = acc2;
}

// ---------------------------------------------------------------------------
// Host launcher
// ---------------------------------------------------------------------------
extern "C" void kernel_launch(void* const* d_in, const int* in_sizes, int n_in,
                              void* d_out, int out_size, void* d_ws, size_t ws_size,
                              hipStream_t stream)
{
    const float* memory     = (const float*)d_in[0];
    const int*   seq_length = (const int*)  d_in[1];
    const int*   tgt        = (const int*)  d_in[2];
    const int*   feat_idx   = (const int*)  d_in[3];
    const float* feat_embs  = (const float*)d_in[4];
    const float* W1         = (const float*)d_in[5];
    const float* b1         = (const float*)d_in[6];
    const float* ln_g       = (const float*)d_in[7];
    const float* ln_b       = (const float*)d_in[8];
    const float* W2         = (const float*)d_in[9];
    const float* b2         = (const float*)d_in[10];
    const float* W3         = (const float*)d_in[11];
    const float* b3         = (const float*)d_in[12];
    const float* Wout       = (const float*)d_in[13];
    const float* leaf_emb   = (const float*)d_in[14];
    const float* leaf_W     = (const float*)d_in[15];
    const float* leaf_b     = (const float*)d_in[16];
    float* out = (float*)d_out;

    // Workspace layout (all offsets 256B-aligned)
    char* ws = (char*)d_ws;
    __bf16* W1t   = (__bf16*)(ws);                        // 768*768*2 = 1179648
    __bf16* W2t   = (__bf16*)(ws + 1179648);
    __bf16* W3t   = (__bf16*)(ws + 2359296);
    __bf16* Woutb = (__bf16*)(ws + 3538944);              // 64*768*2  = 98304
    float*  exm   = (float*) (ws + 3637248);              // 2048*31*4 = 253952
    float*  other = (float*) (ws + 3891200);              // 2048*768*4= 6291456
    // total ~10.2 MB

    const dim3 blk(THREADS);

    // output slots 31..62 are never written by the depth loop -> zero buffer
    hipMemsetAsync(d_out, 0, (size_t)out_size * sizeof(float), stream);

    transpose_to_bf16<<<(768 * 768 + 255) / 256, blk, 0, stream>>>(W1, W1t, DD, DD);
    transpose_to_bf16<<<(768 * 768 + 255) / 256, blk, 0, stream>>>(W2, W2t, DD, DD);
    transpose_to_bf16<<<(768 * 768 + 255) / 256, blk, 0, stream>>>(W3, W3t, DD, DD);
    transpose_to_bf16<<<(VPAD * 768 + 255) / 256, blk, 0, stream>>>(Wout, Woutb, VV, VPAD);
    build_ex<<<(BS + 255) / 256, blk, 0, stream>>>(tgt, seq_length, exm);

    // depth 0 (no other_leaves)
    fused_depth<<<BS / MT, blk, 0, stream>>>(
        feat_embs, feat_idx, memory, other, exm,
        W1t, b1, W2t, b2, W3t, b3, Woutb, ln_g, ln_b, out,
        /*lo=*/0, /*cshift=*/0, /*useOther=*/0);

    for (int d = 1; d < 5; ++d) {
        const int a   = (1 << (d - 1)) - 1;
        const int cp  = 1 << (d - 1);
        build_other<<<BS, blk, 0, stream>>>(tgt, exm, leaf_emb, leaf_W, leaf_b,
                                            other, a, cp);
        const int lo = (1 << d) - 1;
        const int Md = BS << d;
        fused_depth<<<Md / MT, blk, 0, stream>>>(
            feat_embs, feat_idx, memory, other, exm,
            W1t, b1, W2t, b2, W3t, b3, Woutb, ln_g, ln_b, out,
            lo, /*cshift=*/d, /*useOther=*/1);
    }
}
